// SimpleRNN_72868415144449
// MI455X (gfx1250) — matrix operand order repaired and checked by reading, not compiled, not run on hardware
//
#include <hip/hip_runtime.h>
#include <hip/hip_bf16.h>

// ---------------- problem constants ----------------
#define B_ 256
#define T_ 2048
#define I_ 64
#define H_ 256
#define O_ 64
#define F_ 16
#define MB 16                 // batch rows per workgroup (one WMMA M tile)
#define NWAVES 16             // one wave per 16-wide column tile of H
#define THREADS (NWAVES * 32) // 512
#define PAD 8                 // bf16 row padding -> 528B stride (16B aligned, bank-spread)
#define LROW (H_ + PAD)       // 264
#define XROW (I_ + PAD)       // 72

typedef __attribute__((ext_vector_type(16))) __bf16 v16bf;
typedef __attribute__((ext_vector_type(8)))  float  v8f;

union BF16x16 { v16bf v; uint4 q[2]; };
union BF16x2  { __bf16 h[2]; unsigned int u; };

// A-frag (16x32 bf16): lane m = lane%16, half = lane/16.
// elems 0..15 = K[k0+half*16 .. +15]
__device__ __forceinline__ v16bf lds_a_frag(const __bf16* row, int k0, int half) {
  BF16x16 r;
  r.q[0] = *(const uint4*)(row + k0 + half * 16);
  r.q[1] = *(const uint4*)(row + k0 + half * 16 + 8);
  return r.v;
}
// B-frag (32x16 bf16): lane n = lane%16, half = lane/16.
// elems = B[k0+half*16 .. +15][n]; with LDS layout sW[n][k] this is 32 contiguous bytes.
__device__ __forceinline__ v16bf lds_b_frag(const __bf16* row, int k0, int half) {
  BF16x16 r;
  const __bf16* p = row + k0 + half * 16;
  r.q[0] = *(const uint4*)(p);
  r.q[1] = *(const uint4*)(p + 8);
  return r.v;
}
// Build a B-frag directly from 16 contiguous f32 in global memory.
__device__ __forceinline__ v16bf cvt_b_frag(const float* p) {
  v16bf r;
#pragma unroll
  for (int j = 0; j < 16; ++j) r[j] = (__bf16)p[j];
  return r;
}
__device__ __forceinline__ v8f wmma_bf16(v16bf a, v16bf b, v8f c) {
  return __builtin_amdgcn_wmma_f32_16x16x32_bf16(false, a, false, b, (short)0, c, false, false);
}
// Branch-free tanh: v_exp_f32 + v_rcp_f32, no EXEC games in the serial loop.
__device__ __forceinline__ float fast_tanh(float x) {
  float e = __expf(-2.0f * __builtin_fabsf(x));
  float r = __fdividef(1.0f - e, 1.0f + e);
  return __builtin_copysignf(r, x);
}

extern __shared__ char smem_raw[];

// ---------------- fused 2-layer RNN scan ----------------
// grid = B/MB = 16 workgroups, block = 512 (16 waves). Each workgroup owns 16
// batch rows for the full T loop; wave w owns output columns [16w, 16w+16).
// LDS = Wih1 + Whh1 (bf16 [n][k]) + double-buffered h0 + h1 + x = 297,984 B.
// Registers: Whh0 (64 VGPRs) + Wih0 (16 VGPRs) B-fragments, persistent.
// 2 workgroup barriers per timestep.
__global__ __launch_bounds__(THREADS, 1)
void rnn2_fused_kernel(const float* __restrict__ x,
                       const float* __restrict__ hidden,
                       const float* __restrict__ Wih0,
                       const float* __restrict__ Whh0,
                       const float* __restrict__ bih0,
                       const float* __restrict__ bhh0,
                       const float* __restrict__ Wih1,
                       const float* __restrict__ Whh1,
                       const float* __restrict__ bih1,
                       const float* __restrict__ bhh1,
                       __bf16* __restrict__ tail,   // [F][B][H] bf16 ws
                       float* __restrict__ hid_out) // [2][B][H] f32
{
  __bf16* sWih1 = (__bf16*)smem_raw;         // H_ x LROW
  __bf16* sWhh1 = sWih1 + H_ * LROW;         // H_ x LROW
  __bf16* sH0a  = sWhh1 + H_ * LROW;         // MB x LROW (ping)
  __bf16* sH0b  = sH0a + MB * LROW;          // MB x LROW (pong)
  __bf16* sH1   = sH0b + MB * LROW;          // MB x LROW
  __bf16* sX    = sH1 + MB * LROW;           // MB x XROW

  const int tid    = threadIdx.x;
  const int lane   = tid & 31;
  const int wave   = tid >> 5;
  const int lane16 = lane & 15;
  const int half   = lane >> 4;
  const int n0     = wave * 16;
  const int rowBase = blockIdx.x * MB;

  // Stage layer-1 weights to LDS as bf16 (packed cvt + b32 stores)
  for (int idx = tid; idx < H_ * H_ / 2; idx += THREADS) {
    int n = idx >> 7;
    int k = (idx & 127) * 2;
    float2 a = *(const float2*)(Wih1 + (size_t)n * H_ + k);
    float2 b = *(const float2*)(Whh1 + (size_t)n * H_ + k);
    BF16x2 pa; pa.h[0] = (__bf16)a.x; pa.h[1] = (__bf16)a.y;
    BF16x2 pb; pb.h[0] = (__bf16)b.x; pb.h[1] = (__bf16)b.y;
    *(unsigned int*)&sWih1[n * LROW + k] = pa.u;
    *(unsigned int*)&sWhh1[n * LROW + k] = pb.u;
  }
  // Stage initial hidden state (h0 -> ping buffer)
  for (int idx = tid; idx < MB * H_ / 2; idx += THREADS) {
    int r = idx >> 7;
    int k = (idx & 127) * 2;
    float2 a = *(const float2*)(hidden + (size_t)(rowBase + r) * H_ + k);
    float2 b = *(const float2*)(hidden + (size_t)(B_ + rowBase + r) * H_ + k);
    BF16x2 pa; pa.h[0] = (__bf16)a.x; pa.h[1] = (__bf16)a.y;
    BF16x2 pb; pb.h[0] = (__bf16)b.x; pb.h[1] = (__bf16)b.y;
    *(unsigned int*)&sH0a[r * LROW + k] = pa.u;
    *(unsigned int*)&sH1[r * LROW + k] = pb.u;
  }

  // Register-resident layer-0 weight B-fragments
  v16bf whh0f[8];
  {
    const float* wp = Whh0 + (size_t)(n0 + lane16) * H_;
#pragma unroll
    for (int c = 0; c < 8; ++c) whh0f[c] = cvt_b_frag(wp + c * 32 + half * 16);
  }
  v16bf wih0f[2];
  {
    const float* wp = Wih0 + (size_t)(n0 + lane16) * I_;
#pragma unroll
    for (int c = 0; c < 2; ++c) wih0f[c] = cvt_b_frag(wp + c * 32 + half * 16);
  }
  const float bias0 = bih0[n0 + lane16] + bhh0[n0 + lane16];
  const float bias1 = bih1[n0 + lane16] + bhh1[n0 + lane16];

  // Per-thread x staging coordinates; next step's pair kept in registers.
  const int sr = (tid * 2) >> 6;       // row 0..15
  const int si = (tid * 2) & 63;       // col (even)
  const float* xptr = x + (size_t)(rowBase + sr) * T_ * I_ + si;
  unsigned int* sxdst = (unsigned int*)&sX[sr * XROW + si];
  float2 xreg = *(const float2*)xptr;  // t = 0

  // Hot-loop row pointers
  const __bf16* sXrow   = sX + lane16 * XROW;
  const __bf16* sH1row  = sH1 + lane16 * LROW;
  const __bf16* wih1row = sWih1 + (n0 + lane16) * LROW;
  const __bf16* whh1row = sWhh1 + (n0 + lane16) * LROW;
  __bf16* hc = sH0a;  // current h0
  __bf16* hn = sH0b;  // next h0

  __syncthreads();

#pragma unroll 1
  for (int t = 0; t < T_; ++t) {
    // commit this step's x tile from registers (no global latency here)
    {
      BF16x2 p; p.h[0] = (__bf16)xreg.x; p.h[1] = (__bf16)xreg.y;
      *sxdst = p.u;
    }
    __syncthreads();  // B1: sX ready; prev-step h1 (and h0[nxt]) writes visible

    // kick off next step's x load; latency overlaps the whole step
    {
      int tn = (t + 1 < T_) ? (t + 1) : t;
      xreg = *(const float2*)(xptr + (size_t)tn * I_);
    }

    // ---- layer 0: acc0 = x Wih0^T + h0 Whh0^T + b (reads h0[cur]) ----
    v8f acc0;
#pragma unroll
    for (int r = 0; r < 8; ++r) acc0[r] = bias0;
    acc0 = wmma_bf16(lds_a_frag(sXrow, 0, half), wih0f[0], acc0);
    acc0 = wmma_bf16(lds_a_frag(sXrow, 32, half), wih0f[1], acc0);
    {
      const __bf16* hrow = hc + lane16 * LROW;
      v16bf a = lds_a_frag(hrow, 0, half);
#pragma unroll
      for (int c = 0; c < 8; ++c) {
        v16bf an = (c < 7) ? lds_a_frag(hrow, (c + 1) * 32, half) : a;
        acc0 = wmma_bf16(a, whh0f[c], acc0);
        a = an;
      }
    }

    // ---- layer 1 part A: acc1 = h1_old Whh1^T + b (independent of acc0) ----
    v8f acc1;
#pragma unroll
    for (int r = 0; r < 8; ++r) acc1[r] = bias1;
    {
      v16bf a = lds_a_frag(sH1row, 0, half);
      v16bf b = lds_b_frag(whh1row, 0, half);
#pragma unroll
      for (int c = 0; c < 8; ++c) {
        v16bf an = (c < 7) ? lds_a_frag(sH1row, (c + 1) * 32, half) : a;
        v16bf bn = (c < 7) ? lds_b_frag(whh1row, (c + 1) * 32, half) : b;
        acc1 = wmma_bf16(a, b, acc1);
        a = an; b = bn;
      }
    }

    // tanh(h0') and write into the pong buffer (no conflict with h0[cur] readers)
    float h0new[8];
#pragma unroll
    for (int r = 0; r < 8; ++r) h0new[r] = fast_tanh(acc0[r]);
#pragma unroll
    for (int r = 0; r < 8; ++r)
      hn[(r + 8 * half) * LROW + n0 + lane16] = (__bf16)h0new[r];
    __syncthreads();  // B2: h0' visible; all h1_old reads complete

    // ---- layer 1 part B: acc1 += h0' Wih1^T ----
    {
      const __bf16* hrow = hn + lane16 * LROW;
      v16bf a = lds_a_frag(hrow, 0, half);
      v16bf b = lds_b_frag(wih1row, 0, half);
#pragma unroll
      for (int c = 0; c < 8; ++c) {
        v16bf an = (c < 7) ? lds_a_frag(hrow, (c + 1) * 32, half) : a;
        v16bf bn = (c < 7) ? lds_b_frag(wih1row, (c + 1) * 32, half) : b;
        acc1 = wmma_bf16(a, b, acc1);
        a = an; b = bn;
      }
    }
    float h1new[8];
#pragma unroll
    for (int r = 0; r < 8; ++r) h1new[r] = fast_tanh(acc1[r]);
#pragma unroll
    for (int r = 0; r < 8; ++r)
      sH1[(r + 8 * half) * LROW + n0 + lane16] = (__bf16)h1new[r];
    if (t >= T_ - F_) {
      int f = t - (T_ - F_);
#pragma unroll
      for (int r = 0; r < 8; ++r) {
        int m = r + 8 * half;
        tail[((size_t)f * B_ + rowBase + m) * H_ + n0 + lane16] = (__bf16)h1new[r];
      }
    }
    // swap h0 ping/pong; next B1 orders the h1/h0 writes before their readers
    __bf16* tmp = hc; hc = hn; hn = tmp;
  }

  __syncthreads();
  for (int idx = tid; idx < MB * H_; idx += THREADS) {
    int r = idx >> 8, k = idx & 255;
    hid_out[(size_t)(rowBase + r) * H_ + k]      = (float)hc[r * LROW + k];
    hid_out[(size_t)(B_ + rowBase + r) * H_ + k] = (float)sH1[r * LROW + k];
  }
}

// ---------------- FC head: out[b,f,:] = tail[f,b,:] @ fc_W^T + fc_b ----------------
// grid = B, block = 128 (4 waves, one per 16-wide O tile). M tile = the 16 f's of one b.
__global__ __launch_bounds__(128, 1)
void fc_head_kernel(const __bf16* __restrict__ tail,
                    const float* __restrict__ fcW,
                    const float* __restrict__ fcb,
                    float* __restrict__ out)
{
  __bf16* sA = (__bf16*)smem_raw;   // F_ x LROW
  __bf16* sW = sA + F_ * LROW;      // O_ x LROW

  const int tid    = threadIdx.x;
  const int lane   = tid & 31;
  const int wave   = tid >> 5;
  const int lane16 = lane & 15;
  const int half   = lane >> 4;
  const int n0     = wave * 16;
  const int b      = blockIdx.x;

  for (int idx = tid; idx < F_ * H_ / 2; idx += 128) {
    int f = idx >> 7, k = (idx & 127) * 2;
    *(unsigned int*)&sA[f * LROW + k] =
        *(const unsigned int*)&tail[((size_t)f * B_ + b) * H_ + k];
  }
  for (int idx = tid; idx < O_ * H_ / 2; idx += 128) {
    int n = idx >> 7, k = (idx & 127) * 2;
    float2 a = *(const float2*)(fcW + (size_t)n * H_ + k);
    BF16x2 p; p.h[0] = (__bf16)a.x; p.h[1] = (__bf16)a.y;
    *(unsigned int*)&sW[n * LROW + k] = p.u;
  }
  __syncthreads();

  v8f acc;
  const float bias = fcb[n0 + lane16];
#pragma unroll
  for (int r = 0; r < 8; ++r) acc[r] = bias;
  {
    v16bf a = lds_a_frag(sA + lane16 * LROW, 0, half);
    v16bf b = lds_b_frag(sW + (n0 + lane16) * LROW, 0, half);
#pragma unroll
    for (int c = 0; c < 8; ++c) {
      v16bf an = (c < 7) ? lds_a_frag(sA + lane16 * LROW, (c + 1) * 32, half) : a;
      v16bf bn = (c < 7) ? lds_b_frag(sW + (n0 + lane16) * LROW, (c + 1) * 32, half) : b;
      acc = wmma_bf16(a, b, acc);
      a = an; b = bn;
    }
  }
#pragma unroll
  for (int r = 0; r < 8; ++r) {
    int m = r + 8 * half;  // = f
    out[((size_t)b * F_ + m) * O_ + n0 + lane16] = acc[r];
  }
}

extern "C" void kernel_launch(void* const* d_in, const int* in_sizes, int n_in,
                              void* d_out, int out_size, void* d_ws, size_t ws_size,
                              hipStream_t stream) {
  (void)in_sizes; (void)n_in; (void)out_size; (void)ws_size;
  const float* x      = (const float*)d_in[0];
  const float* hidden = (const float*)d_in[1];
  const float* Wih0   = (const float*)d_in[2];
  const float* Whh0   = (const float*)d_in[3];
  const float* bih0   = (const float*)d_in[4];
  const float* bhh0   = (const float*)d_in[5];
  const float* Wih1   = (const float*)d_in[6];
  const float* Whh1   = (const float*)d_in[7];
  const float* bih1   = (const float*)d_in[8];
  const float* bhh1   = (const float*)d_in[9];
  const float* fcW    = (const float*)d_in[10];
  const float* fcb    = (const float*)d_in[11];

  float* out     = (float*)d_out;
  float* hid_out = out + (size_t)B_ * F_ * O_;
  __bf16* tail   = (__bf16*)d_ws;  // F_*B_*H_ bf16 = 2 MB

  const size_t smem_main =
      (size_t)(2 * H_ * LROW + 3 * MB * LROW + MB * XROW) * sizeof(__bf16);
  const size_t smem_fc = (size_t)((F_ + O_) * LROW) * sizeof(__bf16);

  rnn2_fused_kernel<<<dim3(B_ / MB), dim3(THREADS), smem_main, stream>>>(
      x, hidden, Wih0, Whh0, bih0, bhh0, Wih1, Whh1, bih1, bhh1, tail, hid_out);
  fc_head_kernel<<<dim3(B_), dim3(128), smem_fc, stream>>>(tail, fcW, fcb, out);
}
